// GenerativeAttention_3393024164331
// MI455X (gfx1250) — matrix operand order repaired and checked by reading, not compiled, not run on hardware
//
#include <hip/hip_runtime.h>
#include <stdint.h>

// ---------------------------------------------------------------------------
// MHA forward for B=4, S=2048, H=1024, NH=16, HD=64 on gfx1250 (MI455X).
// bf16 WMMA everywhere, fp32 accumulation, flash-attention online softmax.
// GEMMs use double-buffered LDS tiles staged with async global->LDS copies
// (ASYNCcnt), explicit ping-pong so buffer offsets are compile-time.
// ---------------------------------------------------------------------------

typedef __bf16 bf16_t;
typedef __attribute__((ext_vector_type(16))) __bf16 v16bf;
typedef __attribute__((ext_vector_type(8)))  __bf16 v8bf;
typedef __attribute__((ext_vector_type(4)))  __bf16 v4bf;
typedef __attribute__((ext_vector_type(8)))  float  v8f;
typedef __attribute__((ext_vector_type(4)))  float  v4f;
typedef __attribute__((ext_vector_type(4)))  int    v4i;

union FragAB { v16bf v; v8bf h[2]; };

#define WMMA_BF16(a, b, c) \
  __builtin_amdgcn_wmma_f32_16x16x32_bf16(false, (a), false, (b), (short)0, (c), false, false)

static constexpr int Bb = 4;
static constexpr int Ss = 2048;
static constexpr int Hh = 1024;
static constexpr int NHh = 16;
static constexpr int HDd = 64;
static constexpr int Mm = Bb * Ss;   // 8192
static constexpr float NEG = -10000.0f;

// ---- gfx1250 async global->LDS path (guarded; sync fallback otherwise) ----
#if defined(__AMDGCN__) && __has_builtin(__builtin_amdgcn_global_load_async_to_lds_b128)
#define HAS_ASYNC 1
#else
#define HAS_ASYNC 0
#endif

#if HAS_ASYNC
typedef __attribute__((address_space(1))) v4i g_v4i;   // global int4
typedef __attribute__((address_space(3))) v4i l_v4i;   // LDS int4
__device__ __forceinline__ void cp_async_b128(void* lds, const void* g) {
  __builtin_amdgcn_global_load_async_to_lds_b128((g_v4i*)g, (l_v4i*)lds, 0, 0);
}
__device__ __forceinline__ void wait_async0() {
#if __has_builtin(__builtin_amdgcn_s_wait_asynccnt)
  __builtin_amdgcn_s_wait_asynccnt(0);
#else
  asm volatile("s_wait_asynccnt 0x0" ::: "memory");
#endif
}
#endif

// ---------------------------------------------------------------------------
// fp32 -> bf16 conversion (vectorized x4)
// ---------------------------------------------------------------------------
__global__ void mha_cvt_bf16(const float* __restrict__ src, bf16_t* __restrict__ dst, int n4) {
  int i = blockIdx.x * blockDim.x + threadIdx.x;
  if (i < n4) {
    v4f x = *(const v4f*)(src + 4 * (size_t)i);
    v4bf y;
    y.x = (bf16_t)x.x; y.y = (bf16_t)x.y; y.z = (bf16_t)x.z; y.w = (bf16_t)x.w;
    *(v4bf*)(dst + 4 * (size_t)i) = y;
  }
}

// ---------------------------------------------------------------------------
// GEMM: C[M,N] = A[M,K] * W[N,K]^T + bias
//   MODE 0: QKV (z selects weight/bias/dst); bf16 output:
//           z==0 (Q): [B,NH,S,HD], *0.125 (softmax scale folded)
//           z==1 (K): [B,NH,S,HD]
//           z==2 (V): [B,NH,HD,S]  (transposed for PV B-fragments)
//   MODE 1: fp32 row-major output (final projection).
// Block: 128 threads = 4 waves (2x2), tile 128x128, K-step 32.
// Ping-pong LDS buffers; per step: wait async -> barrier -> ds fragment
// loads -> issue next tile's asyncs (overlap WMMAs) -> 16 WMMAs.
// ---------------------------------------------------------------------------
#define G_STR 40   // LDS row stride in halves (80B, 16B-aligned, bank-spread)

template <int MODE>
__global__ __launch_bounds__(128)
void mha_gemm(const bf16_t* __restrict__ A,
              const bf16_t* __restrict__ W0, const bf16_t* __restrict__ W1,
              const bf16_t* __restrict__ W2,
              const float* __restrict__ b0, const float* __restrict__ b1,
              const float* __restrict__ b2,
              bf16_t* __restrict__ q_out, bf16_t* __restrict__ k_out,
              bf16_t* __restrict__ v_out, float* __restrict__ f_out) {
  __shared__ __align__(16) bf16_t As[2][128 * G_STR];
  __shared__ __align__(16) bf16_t Bs[2][128 * G_STR];

  const int tid  = threadIdx.x;
  const int lane = tid & 31;
  const int wave = tid >> 5;
  const int hi   = lane >> 4;
  const int ln   = lane & 15;

  const int n0 = blockIdx.x * 128;
  const int m0 = blockIdx.y * 128;
  const int z  = blockIdx.z;

  const bf16_t* Wm = (MODE == 0) ? (z == 0 ? W0 : (z == 1 ? W1 : W2)) : W0;

  const int wm = (wave >> 1) * 64;  // wave M offset in tile
  const int wn = (wave & 1) * 64;   // wave N offset in tile

  // Per-thread staging chunk coordinates: 512 v8bf chunks per 128x32 tile.
  const int srow = tid >> 2;        // base row (this thread covers srow + 0/32/64/96)
  const int scc  = tid & 3;         // chunk column (8 halves)

  auto stage = [&](int buf, int k0) {
#pragma unroll
    for (int t = 0; t < 4; t++) {
      int row = srow + 32 * t;
      const bf16_t* ga = A  + (size_t)(m0 + row) * Hh + k0 + scc * 8;
      const bf16_t* gb = Wm + (size_t)(n0 + row) * Hh + k0 + scc * 8;
      bf16_t* la = &As[buf][row * G_STR + scc * 8];
      bf16_t* lb = &Bs[buf][row * G_STR + scc * 8];
#if HAS_ASYNC
      cp_async_b128(la, ga);
      cp_async_b128(lb, gb);
#else
      *(v8bf*)la = *(const v8bf*)ga;
      *(v8bf*)lb = *(const v8bf*)gb;
#endif
    }
  };

  v8f acc[4][4];
#pragma unroll
  for (int i = 0; i < 4; i++)
#pragma unroll
    for (int j = 0; j < 4; j++) acc[i][j] = v8f{};

  constexpr int NT = Hh / 32;  // 32 K-steps (even)

  // One pipeline step on a compile-time buffer index.
  auto step = [&](int buf, int next_k0) {
#if HAS_ASYNC
    wait_async0();            // my async writes for this tile are in LDS
#endif
    __syncthreads();          // all writes visible; prior reads of `buf` done

    const bf16_t* as = As[buf];
    const bf16_t* bs = Bs[buf];
    FragAB fa[4], fb[4];
#pragma unroll
    for (int i = 0; i < 4; i++) {
      const bf16_t* p = as + (wm + i * 16 + ln) * G_STR;
      fa[i].h[0] = *(const v8bf*)(p + hi * 8);        // K = {0..7}+8hi
      fa[i].h[1] = *(const v8bf*)(p + 16 + hi * 8);   // K = {16..23}+8hi
    }
#pragma unroll
    for (int j = 0; j < 4; j++) {
      const bf16_t* p = bs + (wn + j * 16 + ln) * G_STR + hi * 8;
      fb[j].h[0] = *(const v8bf*)(p);                 // K = {0..7}+8hi
      fb[j].h[1] = *(const v8bf*)(p + 16);            // K = {16..23}+8hi
    }

    // Issue next tile's copies AFTER the fragment loads so they overlap
    // the WMMAs below even under conservative LDS aliasing.
    if (next_k0 < Hh) stage(buf ^ 1, next_k0);

#pragma unroll
    for (int i = 0; i < 4; i++)
#pragma unroll
      for (int j = 0; j < 4; j++)
        acc[i][j] = WMMA_BF16(fa[i].v, fb[j].v, acc[i][j]);
  };

  stage(0, 0);
#pragma unroll 1
  for (int it = 0; it < NT; it += 2) {
    step(0, (it + 1) * 32);
    step(1, (it + 2) * 32);
  }

  if (MODE == 0) {
    const float* bias = (z == 0 ? b0 : (z == 1 ? b1 : b2));
    bf16_t* dst = (z == 0 ? q_out : (z == 1 ? k_out : v_out));
    const float scale = (z == 0) ? 0.125f : 1.0f;  // 1/sqrt(HD) folded into Q
#pragma unroll
    for (int j = 0; j < 4; j++) {
      int n = n0 + wn + j * 16 + ln;
      float bv = bias[n];
      int head = n >> 6, hd = n & 63;
#pragma unroll
      for (int i = 0; i < 4; i++)
#pragma unroll
        for (int r = 0; r < 8; r++) {
          int m = m0 + wm + i * 16 + hi * 8 + r;   // D row = vgpr + 8*hi
          int b = m >> 11, s = m & 2047;
          float v = (acc[i][j][r] + bv) * scale;
          size_t idx;
          if (z == 2)  // V transposed: [B,NH,HD,S]
            idx = (((size_t)(b * NHh + head) * HDd + hd) << 11) + s;
          else         // Q/K: [B,NH,S,HD]
            idx = (((size_t)(b * NHh + head) * Ss + s) << 6) + hd;
          dst[idx] = (bf16_t)v;
        }
    }
  } else {
#pragma unroll
    for (int j = 0; j < 4; j++) {
      int n = n0 + wn + j * 16 + ln;
      float bv = b0[n];
#pragma unroll
      for (int i = 0; i < 4; i++)
#pragma unroll
        for (int r = 0; r < 8; r++) {
          int m = m0 + wm + i * 16 + hi * 8 + r;
          f_out[(size_t)m * Hh + n] = acc[i][j][r] + bv;
        }
    }
  }
}

// ---------------------------------------------------------------------------
// Flash attention: block = 4 waves, one (b, head, 64-row q block) per block.
// Wave w owns q rows [q0+16w, q0+16w+16). K/V fragments are loaded DIRECTLY
// from global (per-head K/V = 512KB, fully L2-resident; WGP$ absorbs the
// 4-wave reuse) -> no barriers in the main loop. V is pre-transposed
// ([B,NH,HD,S]) so PV B-fragments are contiguous 16B loads. P goes through
// per-wave LDS to convert the f32 D layout into a bf16 A layout.
// ---------------------------------------------------------------------------
#define A_STR 80   // LDS row stride in halves (160B, 16B-aligned)

__global__ __launch_bounds__(128)
void mha_attn(const bf16_t* __restrict__ Q, const bf16_t* __restrict__ Km,
              const bf16_t* __restrict__ Vm, const unsigned char* __restrict__ mask,
              bf16_t* __restrict__ ctx) {
  __shared__ __align__(16) bf16_t Pw[4][16 * A_STR];

  const int tid  = threadIdx.x;
  const int lane = tid & 31;
  const int wave = tid >> 5;
  const int hi   = lane >> 4;
  const int ln   = lane & 15;

  const int q0 = blockIdx.x * 64;
  const int h  = blockIdx.y;
  const int b  = blockIdx.z;

  const size_t bh = ((size_t)b * NHh + h) * Ss * HDd;
  const bf16_t* Qp = Q + bh;    // [s][hd]
  const bf16_t* Kp = Km + bh;   // [key][hd]
  const bf16_t* Vp = Vm + bh;   // [hd][key] (pre-transposed)
  const unsigned char* mp = mask + (size_t)b * Ss;

  // Persistent Q A-fragments (this wave's 16 rows, hd split into 2 chunks).
  FragAB qa[2];
  {
    const bf16_t* qrow = Qp + (size_t)(q0 + wave * 16 + ln) * HDd;
#pragma unroll
    for (int c = 0; c < 2; c++) {
      qa[c].h[0] = *(const v8bf*)(qrow + c * 32 + hi * 8);
      qa[c].h[1] = *(const v8bf*)(qrow + c * 32 + 16 + hi * 8);
    }
  }

  float m_run[8], l_run[8];
  v8f oacc[4];
#pragma unroll
  for (int r = 0; r < 8; r++) { m_run[r] = -1e30f; l_run[r] = 0.0f; }
#pragma unroll
  for (int j = 0; j < 4; j++) oacc[j] = v8f{};

  const int nkt = blockIdx.x + 1;  // causal: key tiles 0..q0/64
  for (int kt = 0; kt < nkt; kt++) {
    const int kb0 = kt * 64;

    // Prefetch next K/V tile into the cache hierarchy (global_prefetch_b8).
    if (kt + 1 < nkt) {
      __builtin_prefetch(Kp + (size_t)(kb0 + 64 + lane) * HDd, 0, 3);
      __builtin_prefetch(Kp + (size_t)(kb0 + 96 + lane) * HDd, 0, 3);
      __builtin_prefetch(Vp + (size_t)lane * Ss + kb0 + 64, 0, 3);
      __builtin_prefetch(Vp + (size_t)(lane + 32) * Ss + kb0 + 64, 0, 3);
    }

    // Scores: S = Q * K^T  (Q pre-scaled by 1/sqrt(HD)); K straight from global.
    v8f sacc[4];
#pragma unroll
    for (int nt = 0; nt < 4; nt++) {
      const bf16_t* kr = Kp + (size_t)(kb0 + nt * 16 + ln) * HDd;
      FragAB kf0, kf1;
      kf0.h[0] = *(const v8bf*)(kr + hi * 8);
      kf0.h[1] = *(const v8bf*)(kr + hi * 8 + 16);
      kf1.h[0] = *(const v8bf*)(kr + 32 + hi * 8);
      kf1.h[1] = *(const v8bf*)(kr + 32 + hi * 8 + 16);
      v8f a = v8f{};
      a = WMMA_BF16(qa[0].v, kf0.v, a);
      a = WMMA_BF16(qa[1].v, kf1.v, a);
      sacc[nt] = a;
    }

    // Padding + causal mask (diagonal tile only).
    const bool diag = (kt == nkt - 1);
#pragma unroll
    for (int nt = 0; nt < 4; nt++) {
      const bool pad = (mp[kb0 + nt * 16 + ln] == 0);
#pragma unroll
      for (int r = 0; r < 8; r++) {
        float s = sacc[nt][r];
        if (pad) s = NEG;
        if (diag) {
          int qrow = q0 + wave * 16 + hi * 8 + r;
          if (kb0 + nt * 16 + ln > qrow) s = NEG;
        }
        sacc[nt][r] = s;
      }
    }

    // Online softmax per row (rows replicated across each 16-lane half).
#pragma unroll
    for (int r = 0; r < 8; r++) {
      float rm = fmaxf(fmaxf(sacc[0][r], sacc[1][r]),
                       fmaxf(sacc[2][r], sacc[3][r]));
      rm = fmaxf(rm, __shfl_xor(rm, 1));
      rm = fmaxf(rm, __shfl_xor(rm, 2));
      rm = fmaxf(rm, __shfl_xor(rm, 4));
      rm = fmaxf(rm, __shfl_xor(rm, 8));
      float mnew = fmaxf(m_run[r], rm);
      float scl  = __expf(m_run[r] - mnew);
      float rs = 0.0f;
#pragma unroll
      for (int nt = 0; nt < 4; nt++) {
        float p = __expf(sacc[nt][r] - mnew);
        sacc[nt][r] = p;
        rs += p;
      }
      rs += __shfl_xor(rs, 1);
      rs += __shfl_xor(rs, 2);
      rs += __shfl_xor(rs, 4);
      rs += __shfl_xor(rs, 8);
      l_run[r] = l_run[r] * scl + rs;
      m_run[r] = mnew;
#pragma unroll
      for (int j = 0; j < 4; j++) oacc[j][r] *= scl;
    }

    // P (D layout) -> per-wave LDS -> A-fragment layout (bf16).
    bf16_t* P = &Pw[wave][0];
#pragma unroll
    for (int nt = 0; nt < 4; nt++)
#pragma unroll
      for (int r = 0; r < 8; r++)
        P[(hi * 8 + r) * A_STR + nt * 16 + ln] = (bf16_t)sacc[nt][r];

    FragAB pa[2];
#pragma unroll
    for (int kk = 0; kk < 2; kk++) {
      const bf16_t* pr = P + ln * A_STR + kk * 32;
      pa[kk].h[0] = *(const v8bf*)(pr + hi * 8);
      pa[kk].h[1] = *(const v8bf*)(pr + 16 + hi * 8);
    }

    // O += P * V ; V^T fragments straight from global (lane = hd row,
    // halves = contiguous keys).
#pragma unroll
    for (int kk = 0; kk < 2; kk++)
#pragma unroll
      for (int j = 0; j < 4; j++) {
        const bf16_t* vr = Vp + (size_t)(j * 16 + ln) * Ss + kb0 + kk * 32 + hi * 8;
        FragAB vf;
        vf.h[0] = *(const v8bf*)(vr);
        vf.h[1] = *(const v8bf*)(vr + 16);
        oacc[j] = WMMA_BF16(pa[kk].v, vf.v, oacc[j]);
      }
  }

  // Normalize and write ctx[b, s, head*64 + hd] (bf16).
#pragma unroll
  for (int r = 0; r < 8; r++) {
    float inv = 1.0f / l_run[r];
    int row = q0 + wave * 16 + hi * 8 + r;
    size_t base = ((size_t)b * Ss + row) * Hh + h * HDd;
#pragma unroll
    for (int j = 0; j < 4; j++)
      ctx[base + j * 16 + ln] = (bf16_t)(oacc[j][r] * inv);
  }
}

// ---------------------------------------------------------------------------
// Host-side launcher. Workspace layout (88 MB total):
//   Xb 16MB | Wq/Wk/Wv/Wp bf16 2MB ea | Qb/Kb/Vb 16MB ea | Ctx 16MB
// ---------------------------------------------------------------------------
extern "C" void kernel_launch(void* const* d_in, const int* in_sizes, int n_in,
                              void* d_out, int out_size, void* d_ws, size_t ws_size,
                              hipStream_t stream) {
  const float* x  = (const float*)d_in[0];
  const unsigned char* mask = (const unsigned char*)d_in[1];
  const float* Wq = (const float*)d_in[2];
  const float* bq = (const float*)d_in[3];
  const float* Wk = (const float*)d_in[4];
  const float* bk = (const float*)d_in[5];
  const float* Wv = (const float*)d_in[6];
  const float* bv = (const float*)d_in[7];
  const float* Wp = (const float*)d_in[8];
  const float* bp = (const float*)d_in[9];
  float* out = (float*)d_out;

  char* ws = (char*)d_ws;
  bf16_t* Xb  = (bf16_t*)ws; ws += (size_t)Mm * Hh * 2;
  bf16_t* Wqb = (bf16_t*)ws; ws += (size_t)Hh * Hh * 2;
  bf16_t* Wkb = (bf16_t*)ws; ws += (size_t)Hh * Hh * 2;
  bf16_t* Wvb = (bf16_t*)ws; ws += (size_t)Hh * Hh * 2;
  bf16_t* Wpb = (bf16_t*)ws; ws += (size_t)Hh * Hh * 2;
  bf16_t* Qb  = (bf16_t*)ws; ws += (size_t)Mm * Hh * 2;
  bf16_t* Kb  = (bf16_t*)ws; ws += (size_t)Mm * Hh * 2;
  bf16_t* Vb  = (bf16_t*)ws; ws += (size_t)Mm * Hh * 2;  // transposed [B,NH,HD,S]
  bf16_t* Ctx = (bf16_t*)ws; ws += (size_t)Mm * Hh * 2;

  // fp32 -> bf16 conversions
  {
    int n4 = (Mm * Hh) / 4;  // 2,097,152
    mha_cvt_bf16<<<n4 / 256, 256, 0, stream>>>(x, Xb, n4);
    int w4 = (Hh * Hh) / 4;  // 262,144
    mha_cvt_bf16<<<w4 / 256, 256, 0, stream>>>(Wq, Wqb, w4);
    mha_cvt_bf16<<<w4 / 256, 256, 0, stream>>>(Wk, Wkb, w4);
    mha_cvt_bf16<<<w4 / 256, 256, 0, stream>>>(Wv, Wvb, w4);
    mha_cvt_bf16<<<w4 / 256, 256, 0, stream>>>(Wp, Wpb, w4);
  }

  // QKV projections: grid (N/128, M/128, 3)
  mha_gemm<0><<<dim3(Hh / 128, Mm / 128, 3), 128, 0, stream>>>(
      Xb, Wqb, Wkb, Wvb, bq, bk, bv, Qb, Kb, Vb, nullptr);

  // Flash attention: grid (S/64, NH, B)
  mha_attn<<<dim3(Ss / 64, NHh, Bb), 128, 0, stream>>>(Qb, Kb, Vb, mask, Ctx);

  // Output projection: fp32 out + bias
  mha_gemm<1><<<dim3(Hh / 128, Mm / 128, 1), 128, 0, stream>>>(
      Ctx, Wpb, nullptr, nullptr, bp, nullptr, nullptr,
      nullptr, nullptr, nullptr, out);
}
